// STGCNDynamic_33767032881622
// MI455X (gfx1250) — matrix operand order repaired and checked by reading, not compiled, not run on hardware
//
#include <hip/hip_runtime.h>
#include <hip/hip_bf16.h>

// ---------------------------------------------------------------------------
// Types / WMMA helpers (CDNA5 wave32, v_wmma_f32_16x16x32_f16)
// ---------------------------------------------------------------------------
typedef __attribute__((ext_vector_type(16))) _Float16 v16h;
typedef __attribute__((ext_vector_type(8)))  _Float16 v8h;
typedef __attribute__((ext_vector_type(8)))  float    v8f;

#define NNODES 23
#define BB     1024

__device__ __forceinline__ v16h cat16(v8h a, v8h b) {
    return __builtin_shufflevector(a, b, 0,1,2,3,4,5,6,7,8,9,10,11,12,13,14,15);
}

// A fragment (16x32 f16): lane l holds row m=l&15; element i maps to
// K = (l>>4)*8 + i          (i=0..7)
// K = 16 + (l>>4)*8 + (i-8) (i=8..15)
// rowBase = &W[m*KP + k0]  (16B aligned runs)
__device__ __forceinline__ v16h load_afrag(const _Float16* rowBase, int lhalf) {
    v8h a0 = *(const v8h*)(rowBase + lhalf * 8);
    v8h a1 = *(const v8h*)(rowBase + 16 + lhalf * 8);
    return cat16(a0, a1);
}

// B fragment (32x16 f16) from an LDS panel stored [col][k]:
// lane l = column (l&15); K = (l>>4)*8 + i (i<8), 16 + (l>>4)*8 + (i-8) (i>=8).
__device__ __forceinline__ v16h load_bfrag(const _Float16* colBase /* = &panel[col*stride + k0] */,
                                           int lhalf) {
    v8h b0 = *(const v8h*)(colBase + lhalf * 8);
    v8h b1 = *(const v8h*)(colBase + 16 + lhalf * 8);
    return cat16(b0, b1);
}

__device__ __forceinline__ v8f wmma_f16(v16h a, v16h b, v8f c) {
    return __builtin_amdgcn_wmma_f32_16x16x32_f16(false, a, false, b, (short)0, c, false, false);
}

// ---------------------------------------------------------------------------
// Weight packing: conv weight (M, Cin, Kt, 1) f32 -> f16 [M][KP], zero padded
// (Kt==1 handles the 1x1 conv matrices too)
// ---------------------------------------------------------------------------
__global__ void prep_convw_kernel(const float* __restrict__ src, _Float16* __restrict__ dst,
                                  int M, int Cin, int Kt, int KP) {
    int idx = blockIdx.x * 256 + threadIdx.x;
    if (idx >= M * KP) return;
    int m = idx / KP, kk = idx - m * KP;
    float v = 0.f;
    if (kk < Cin * Kt) {
        int ci = kk / Kt, dt = kk - ci * Kt;
        v = src[(size_t)((m * Cin + ci) * Kt + dt)];
    }
    dst[idx] = (_Float16)v;
}

// th (3, Cin, Cout) f32 -> f16 [3][Cout][Cp] (transposed, padded)
__global__ void prep_th_kernel(const float* __restrict__ src, _Float16* __restrict__ dst,
                               int Cin, int Cout, int Cp) {
    int idx = blockIdx.x * 256 + threadIdx.x;
    if (idx >= 3 * Cout * Cp) return;
    int k = idx / (Cout * Cp);
    int rem = idx - k * (Cout * Cp);
    int o = rem / Cp, c = rem - o * Cp;
    float v = (c < Cin) ? src[(size_t)((k * Cin + c) * Cout + o)] : 0.f;
    dst[idx] = (_Float16)v;
}

// x (B,50,23,2) f32 -> act0 f16 [c][t][n][b]
__global__ void transpose_in_kernel(const float* __restrict__ x, _Float16* __restrict__ a0) {
    int idx = blockIdx.x * 256 + threadIdx.x;
    if (idx >= 2 * 50 * NNODES * BB) return;
    int b = idx & (BB - 1);
    int r = idx >> 10;               // ((c*50+t)*23+n)
    int n = r % NNODES; r /= NNODES;
    int t = r % 50;     int c = r / 50;
    a0[idx] = (_Float16)x[(size_t)(((b * 50 + t) * NNODES + n) * 2 + c)];
}

// ---------------------------------------------------------------------------
// Chebyshev basis per batch element. Lkh f16 layout [k][i][j][b].
// lmax(eigvalsh) replaced by power iteration (L symmetric PSD).
// ---------------------------------------------------------------------------
__global__ __launch_bounds__(64) void cheb_kernel(const float* __restrict__ x,
                                                  _Float16* __restrict__ Lkh) {
    __shared__ float px[NNODES], py[NNODES];
    __shared__ float Wm[NNODES * NNODES], Lm[NNODES * NNODES];
    __shared__ float vv0[NNODES], vv1[NNODES];
    __shared__ float red[64];
    __shared__ float scal;
    const int b = blockIdx.x, tid = threadIdx.x;

    // mean position over 50 timesteps
    for (int p = tid; p < NNODES * 2; p += 64) {
        int n = p >> 1, d = p & 1;
        float s = 0.f;
        for (int tt = 0; tt < 50; ++tt)
            s += x[(size_t)(((b * 50 + tt) * NNODES + n) * 2 + d)];
        s *= (1.0f / 50.0f);
        if (d) py[n] = s; else px[n] = s;
    }
    __syncthreads();

    // pairwise distances + sigma
    float part = 0.f;
    for (int e = tid; e < NNODES * NNODES; e += 64) {
        int i = e / NNODES, jn = e - i * NNODES;
        float d = 0.f;
        if (i != jn) { float dx = px[i] - px[jn], dy = py[i] - py[jn]; d = sqrtf(dx * dx + dy * dy); }
        Wm[e] = d; part += d;
    }
    red[tid] = part; __syncthreads();
    if (tid == 0) {
        float s = 0.f;
        for (int q = 0; q < 64; ++q) s += red[q];
        float sig = s / (float)(NNODES * (NNODES - 1));
        scal = fmaxf(sig * sig, 1e-6f);
    }
    __syncthreads();
    const float sigma2 = scal;

    // signed gaussian adjacency
    for (int e = tid; e < NNODES * NNODES; e += 64) {
        int i = e / NNODES, jn = e - i * NNODES;
        float ti = (i >= 11 && i < 22) ? 1.f : ((i == 22) ? -1.f : 0.f);
        float tj = (jn >= 11 && jn < 22) ? 1.f : ((jn == 22) ? -1.f : 0.f);
        bool ball = (ti == -1.f) || (tj == -1.f);
        bool riv = (ti != tj) && !ball;
        float sgn = riv ? -1.f : 1.f;
        float d = Wm[e];
        Wm[e] = (i == jn) ? 0.f : __expf(-d * d / sigma2) * sgn;
    }
    __syncthreads();

    // normalized Laplacian (dinv into px)
    for (int i = tid; i < NNODES; i += 64) {
        float s = 0.f;
        for (int jn = 0; jn < NNODES; ++jn) s += fabsf(Wm[i * NNODES + jn]);
        px[i] = rsqrtf(s + 1e-8f);
    }
    __syncthreads();
    for (int e = tid; e < NNODES * NNODES; e += 64) {
        int i = e / NNODES, jn = e - i * NNODES;
        Lm[e] = ((i == jn) ? 1.f : 0.f) - px[i] * Wm[e] * px[jn];
    }
    for (int i = tid; i < NNODES; i += 64) vv0[i] = 1.0f + 0.01f * (float)i;
    __syncthreads();

    // power iteration for lmax
    float nrm = 1.f;
    for (int it = 0; it < 50; ++it) {
        for (int i = tid; i < NNODES; i += 64) {
            float s = 0.f;
            for (int jn = 0; jn < NNODES; ++jn) s += Lm[i * NNODES + jn] * vv0[jn];
            vv1[i] = s;
        }
        __syncthreads();
        if (tid == 0) {
            float s = 0.f;
            for (int q = 0; q < NNODES; ++q) s += vv1[q] * vv1[q];
            scal = sqrtf(s);
        }
        __syncthreads();
        nrm = scal;
        float rr = (nrm > 1e-20f) ? 1.f / nrm : 0.f;
        for (int i = tid; i < NNODES; i += 64) vv0[i] = vv1[i] * rr;
        __syncthreads();
    }
    const float lmax = fmaxf(nrm, 1e-6f);

    // Lt (into Wm), then T2 = 2*Lt*Lt - I ; store all three polys
    for (int e = tid; e < NNODES * NNODES; e += 64) {
        int i = e / NNODES, jn = e - i * NNODES;
        Wm[e] = 2.f * Lm[e] / lmax - ((i == jn) ? 1.f : 0.f);
    }
    __syncthreads();
    for (int e = tid; e < NNODES * NNODES; e += 64) {
        int i = e / NNODES, jn = e - i * NNODES;
        float t2 = 0.f;
        for (int m2 = 0; m2 < NNODES; ++m2) t2 += Wm[i * NNODES + m2] * Wm[m2 * NNODES + jn];
        t2 = 2.f * t2 - ((i == jn) ? 1.f : 0.f);
        Lkh[(size_t)(0 * NNODES * NNODES + e) * BB + b] = (_Float16)((i == jn) ? 1.f : 0.f);
        Lkh[(size_t)(1 * NNODES * NNODES + e) * BB + b] = (_Float16)Wm[e];
        Lkh[(size_t)(2 * NNODES * NNODES + e) * BB + b] = (_Float16)t2;
    }
}

// ---------------------------------------------------------------------------
// Temporal conv + GLU (+optional 1x1-conv residual), WMMA GEMM.
//   y = W[2*Cout x Cin*Kt] * patch ; out = (y1 + bias1 + res) * sigmoid(y2 + bias2)
// Block: 32*(Cout/16) threads; wave w owns channel tile [w*16, w*16+16).
// B panel (K x 16 batch cols) staged in LDS as [col][k]; residual tap rows
// staged into a second contiguous panel sR[col][ci] so the residual GEMM uses
// the same branch-free b128 fragment loads as the main loop.
// ---------------------------------------------------------------------------
__global__ __launch_bounds__(256) void tconv_glu_kernel(
    const _Float16* __restrict__ Wt, int KP,
    const _Float16* __restrict__ At, int Cp,
    const float* __restrict__ bias,
    const _Float16* __restrict__ xin,
    _Float16* __restrict__ yout,
    int Cin, int Cout, int Kt, int Tin, int Tout, int ciChunk)
{
    __shared__ _Float16 sB[16 * 1344];
    __shared__ _Float16 sR[16 * 128];
    const int b0 = blockIdx.x * 16;
    const int tn = blockIdx.y;
    const int t = tn / NNODES, n = tn - t * NNODES;
    const int tid = threadIdx.x;
    const int lane = tid & 31, wave = tid >> 5;
    const int lhalf = lane >> 4, lcol = lane & 15;
    const int mwaves = Cout >> 4;
    const int co0 = wave * 16;

    v8f acc1 = {}, acc2 = {}, accR = {};

    for (int ci0 = 0; ci0 < Cin; ci0 += ciChunk) {
        const int ciCnt = (ciChunk < Cin - ci0) ? ciChunk : (Cin - ci0);
        const int cpad = (ciCnt + 31) & ~31;
        const int Kc = ciCnt * Kt;
        const int Kcp = (Kc + 31) & ~31;
        // ---- stage activation panel into LDS ([col][k]) ----
        for (int r = tid; r < Kcp; r += blockDim.x) {
            if (r < Kc) {
                const int ciL = r / Kt, dt = r - ciL * Kt;
                const int ci = ci0 + ciL;
                const _Float16* src = xin + (size_t)(((ci * Tin + t + dt) * NNODES + n) * BB + b0);
                v8h lo = *(const v8h*)src;
                v8h hi = *(const v8h*)(src + 8);
#pragma unroll
                for (int c = 0; c < 8; ++c) { sB[c * Kcp + r] = lo[c]; sB[(c + 8) * Kcp + r] = hi[c]; }
            } else {
#pragma unroll
                for (int c = 0; c < 16; ++c) sB[c * Kcp + r] = (_Float16)0;
            }
        }
        // ---- stage residual tap rows into sR ([col][ci]) ----
        if (At) {
            for (int r = tid; r < cpad; r += blockDim.x) {
                if (r < ciCnt) {
                    const int ci = ci0 + r;
                    const _Float16* src = xin + (size_t)(((ci * Tin + t + Kt - 1) * NNODES + n) * BB + b0);
                    v8h lo = *(const v8h*)src;
                    v8h hi = *(const v8h*)(src + 8);
#pragma unroll
                    for (int c = 0; c < 8; ++c) { sR[c * cpad + r] = lo[c]; sR[(c + 8) * cpad + r] = hi[c]; }
                } else {
#pragma unroll
                    for (int c = 0; c < 16; ++c) sR[c * cpad + r] = (_Float16)0;
                }
            }
        }
        __syncthreads();
        if (wave < mwaves) {
            const int gk0 = ci0 * Kt;
            const _Float16* w1row = Wt + (size_t)(co0 + lcol) * KP + gk0;
            const _Float16* w2row = Wt + (size_t)(Cout + co0 + lcol) * KP + gk0;
            const _Float16* sbrow = sB + lcol * Kcp;
            for (int ks = 0; ks < Kcp; ks += 32) {
                v16h bf = load_bfrag(sbrow + ks, lhalf);
                v16h a1 = load_afrag(w1row + ks, lhalf);
                acc1 = wmma_f16(a1, bf, acc1);
                v16h a2 = load_afrag(w2row + ks, lhalf);
                acc2 = wmma_f16(a2, bf, acc2);
            }
            if (At) {  // residual = (A * x)[t + Kt - 1], branch-free fragments
                const _Float16* arow = At + (size_t)(co0 + lcol) * Cp + ci0;
                const _Float16* srrow = sR + lcol * cpad;
                for (int cs = 0; cs < cpad; cs += 32) {
                    v16h bf = load_bfrag(srrow + cs, lhalf);
                    v16h aR = load_afrag(arow + cs, lhalf);
                    accR = wmma_f16(aR, bf, accR);
                }
            }
        }
        __syncthreads();
    }

    if (wave < mwaves) {
#pragma unroll
        for (int r = 0; r < 8; ++r) {
            const int co = co0 + r + lhalf * 8;
            const int b = b0 + lcol;
            float res;
            if (At) res = accR[r];
            else    res = (float)xin[(size_t)(((co * Tin + t + Kt - 1) * NNODES + n) * BB + b)];
            float y1 = acc1[r] + bias[co] + res;
            float y2 = acc2[r] + bias[Cout + co];
            float o = y1 * (1.0f / (1.0f + __expf(-y2)));
            yout[(size_t)(((co * Tout + t) * NNODES + n) * BB + b)] = (_Float16)o;
        }
    }
}

// ---------------------------------------------------------------------------
// sconv step 1: u_k = th_k^T (Cout x Cin) * x  (shared GEMM over all b) — WMMA
// uout layout [3][Cout][Tc][NNODES][BB]
// ---------------------------------------------------------------------------
__global__ __launch_bounds__(128) void theta_gemm_kernel(
    const _Float16* __restrict__ ThT, // [3][Cout][Cp]
    const _Float16* __restrict__ xin, // [Cin][T][NNODES][BB]
    _Float16* __restrict__ uout,
    int Cin, int Cout, int Cp, int T, int t0, int Tc)
{
    __shared__ _Float16 sB[16 * 128];
    const int b0 = blockIdx.x * 16;
    const int tjj = blockIdx.y;
    const int tl = tjj / NNODES, j = tjj - tl * NNODES;
    const int t = t0 + tl;
    const int k = blockIdx.z;
    const int tid = threadIdx.x;
    const int lane = tid & 31, wave = tid >> 5;
    const int lhalf = lane >> 4, lcol = lane & 15;

    for (int r = tid; r < Cp; r += blockDim.x) {
        if (r < Cin) {
            const _Float16* src = xin + (size_t)(((r * T + t) * NNODES + j) * BB) + b0;
            v8h lo = *(const v8h*)src;
            v8h hi = *(const v8h*)(src + 8);
#pragma unroll
            for (int c = 0; c < 8; ++c) { sB[c * Cp + r] = lo[c]; sB[(c + 8) * Cp + r] = hi[c]; }
        } else {
#pragma unroll
            for (int c = 0; c < 16; ++c) sB[c * Cp + r] = (_Float16)0;
        }
    }
    __syncthreads();
    const int mwaves = Cout >> 4;
    if (wave >= mwaves) return;
    const int co0 = wave * 16;
    v8f acc = {};
    const _Float16* arow = ThT + (size_t)(k * Cout + co0 + lcol) * Cp;
    const _Float16* sbrow = sB + lcol * Cp;
    for (int ks = 0; ks < Cp; ks += 32) {
        v16h bf = load_bfrag(sbrow + ks, lhalf);
        v16h af = load_afrag(arow + ks, lhalf);
        acc = wmma_f16(af, bf, acc);
    }
#pragma unroll
    for (int r = 0; r < 8; ++r) {
        int co = co0 + r + lhalf * 8;
        int b = b0 + lcol;
        uout[(size_t)(((k * Cout + co) * Tc + tl) * NNODES + j) * BB + b] = (_Float16)acc[r];
    }
}

// ---------------------------------------------------------------------------
// sconv step 2: y[o,t,i,b] = relu( sum_{k,j} Lk[b,k,i,j] * u_k[o,t,j,b] + sb[o] + x[o,t,i,b] )
// Lk per-column (batch) matrices -> VALU pass, Lk tile staged in LDS.
// ---------------------------------------------------------------------------
__global__ __launch_bounds__(256) void sconv_step2_kernel(
    const _Float16* __restrict__ u,
    const _Float16* __restrict__ Lkh, // [3][NNODES][NNODES][BB]
    const _Float16* __restrict__ xin,
    const float* __restrict__ sb,
    _Float16* __restrict__ yout,
    int C, int T, int t0, int Tc)
{
    __shared__ _Float16 sL[3 * NNODES * NNODES * 16];
    const int tl = blockIdx.x;
    const int b0 = blockIdx.y * 16;
    for (int e = threadIdx.x; e < 3 * NNODES * NNODES * 16; e += 256) {
        int kij = e >> 4, bl = e & 15;
        sL[e] = Lkh[(size_t)kij * BB + b0 + bl];
    }
    __syncthreads();
    const int bl = threadIdx.x & 15;
    const int wk = threadIdx.x >> 4;
    const int t = t0 + tl;
    for (int oi = wk; oi < C * NNODES; oi += 16) {
        int o = oi / NNODES, i = oi - o * NNODES;
        float acc = sb[o];
        for (int k = 0; k < 3; ++k) {
            const _Float16* uk = u + (size_t)(((k * C + o) * Tc + tl) * NNODES) * BB + b0 + bl;
            const _Float16* lr = sL + (size_t)((k * NNODES + i) * NNODES) * 16 + bl;
#pragma unroll
            for (int jn = 0; jn < NNODES; ++jn)
                acc += (float)lr[jn * 16] * (float)uk[(size_t)jn * BB];
        }
        acc += (float)xin[(size_t)(((o * T + t) * NNODES + i) * BB) + b0 + bl];
        acc = fmaxf(acc, 0.f);
        yout[(size_t)(((o * T + t) * NNODES + i) * BB) + b0 + bl] = (_Float16)acc;
    }
}

// ---------------------------------------------------------------------------
// LayerNorm over (node, channel) per (b, t); in place; thread = batch index.
// ---------------------------------------------------------------------------
__global__ __launch_bounds__(1024) void ln_kernel(_Float16* __restrict__ act,
                                                  const float* __restrict__ g,
                                                  const float* __restrict__ bt,
                                                  int C, int T)
{
    const int t = blockIdx.x;
    const int b = threadIdx.x;
    float s = 0.f, s2 = 0.f;
    for (int n = 0; n < NNODES; ++n)
        for (int c = 0; c < C; ++c) {
            float v = (float)act[(size_t)(((c * T + t) * NNODES + n) * BB) + b];
            s += v; s2 += v * v;
        }
    const float inv = 1.0f / (float)(NNODES * C);
    const float m = s * inv;
    const float var = s2 * inv - m * m;
    const float rs = rsqrtf(var + 1e-5f);
    for (int n = 0; n < NNODES; ++n)
        for (int c = 0; c < C; ++c) {
            size_t idx = (size_t)(((c * T + t) * NNODES + n) * BB) + b;
            float v = (float)act[idx];
            act[idx] = (_Float16)((v - m) * rs * g[n * C + c] + bt[n * C + c]);
        }
}

// ---------------------------------------------------------------------------
// Head: s = sigmoid(W2*h + b2); out = FW*s + fb; reorder to (B,10,23,2) f32.
// ---------------------------------------------------------------------------
__global__ __launch_bounds__(256) void final_kernel(
    const _Float16* __restrict__ h,   // [128][1][NNODES][BB]
    const _Float16* __restrict__ W2,  // [128][128]
    const float* __restrict__ b2,
    const _Float16* __restrict__ FW,  // [20][128]
    const float* __restrict__ fb,
    float* __restrict__ out)
{
    __shared__ _Float16 sW[128 * 128];
    __shared__ _Float16 sF[20 * 128];
    const int n = blockIdx.x;
    const int b = blockIdx.y * 256 + threadIdx.x;
    for (int e = threadIdx.x; e < 128 * 128; e += 256) sW[e] = W2[e];
    for (int e = threadIdx.x; e < 20 * 128; e += 256) sF[e] = FW[e];
    __syncthreads();
    float hreg[128];
#pragma unroll
    for (int c = 0; c < 128; ++c)
        hreg[c] = (float)h[(size_t)((c * NNODES + n) * BB) + b];
    float acc[20];
#pragma unroll
    for (int p = 0; p < 20; ++p) acc[p] = 0.f;
    for (int c2 = 0; c2 < 128; ++c2) {
        float d = b2[c2];
#pragma unroll
        for (int c = 0; c < 128; ++c) d += (float)sW[c2 * 128 + c] * hreg[c];
        float sg = 1.f / (1.f + __expf(-d));
#pragma unroll
        for (int p = 0; p < 20; ++p) acc[p] += (float)sF[p * 128 + c2] * sg;
    }
#pragma unroll
    for (int q = 0; q < 10; ++q)
#pragma unroll
        for (int d2 = 0; d2 < 2; ++d2)
            out[(size_t)(((b * 10 + q) * NNODES + n) * 2 + d2)] = acc[q * 2 + d2] + fb[q * 2 + d2];
}

// ---------------------------------------------------------------------------
// Host driver
// ---------------------------------------------------------------------------
extern "C" void kernel_launch(void* const* d_in, const int* in_sizes, int n_in,
                              void* d_out, int out_size, void* d_ws, size_t ws_size,
                              hipStream_t stream) {
    (void)in_sizes; (void)n_in; (void)out_size; (void)ws_size;
    const float* x       = (const float*)d_in[0];
    const float* b1_t1w  = (const float*)d_in[1];
    const float* b1_t1b  = (const float*)d_in[2];
    const float* b1_t1a  = (const float*)d_in[3];
    const float* b1_th   = (const float*)d_in[4];
    const float* b1_sb   = (const float*)d_in[5];
    const float* b1_t2w  = (const float*)d_in[6];
    const float* b1_t2b  = (const float*)d_in[7];
    const float* b1_t2a  = (const float*)d_in[8];
    const float* b1_g    = (const float*)d_in[9];
    const float* b1_bb   = (const float*)d_in[10];
    const float* b2_t1w  = (const float*)d_in[11];
    const float* b2_t1b  = (const float*)d_in[12];
    const float* b2_th   = (const float*)d_in[13];
    const float* b2_sb   = (const float*)d_in[14];
    const float* b2_t2w  = (const float*)d_in[15];
    const float* b2_t2b  = (const float*)d_in[16];
    const float* b2_t2a  = (const float*)d_in[17];
    const float* b2_g    = (const float*)d_in[18];
    const float* b2_bb   = (const float*)d_in[19];
    const float* o_t1w   = (const float*)d_in[20];
    const float* o_t1b   = (const float*)d_in[21];
    const float* o_g     = (const float*)d_in[22];
    const float* o_bb    = (const float*)d_in[23];
    const float* o_t2w   = (const float*)d_in[24];
    const float* o_t2b   = (const float*)d_in[25];
    const float* o_fw    = (const float*)d_in[26];
    const float* o_fb    = (const float*)d_in[27];

    _Float16* base = (_Float16*)d_ws;
    size_t off = 0;
    auto alloc = [&](size_t nh) { _Float16* p = base + off; off += (nh + 63) & ~(size_t)63; return p; };

    _Float16* wB1T1 = alloc(64 * 32);
    _Float16* wB1T2 = alloc(128 * 96);
    _Float16* wB2T1 = alloc(128 * 192);
    _Float16* wB2T2 = alloc(256 * 192);
    _Float16* wOT1  = alloc((size_t)256 * 5376);
    _Float16* aB1T1 = alloc(32 * 32);
    _Float16* aB1T2 = alloc(64 * 32);
    _Float16* aB2T2 = alloc(128 * 64);
    _Float16* wOT2  = alloc(128 * 128);
    _Float16* wOF   = alloc(20 * 128);
    _Float16* th1   = alloc(3 * 32 * 32);
    _Float16* th2   = alloc(3 * 64 * 64);
    _Float16* A0    = alloc((size_t)2 * 50 * NNODES * BB);
    _Float16* Lk    = alloc((size_t)3 * NNODES * NNODES * BB);
    _Float16* arA   = alloc((size_t)64 * 46 * NNODES * BB);    // max of h3/h5/h7
    _Float16* arB   = alloc((size_t)128 * 42 * NNODES * BB);   // max of h2/h4/h6
    _Float16* uA    = alloc((size_t)3 * 64 * 12 * NNODES * BB);

    auto g1 = [](int total) { return dim3((unsigned)((total + 255) / 256)); };

    // ---- weight packing ----
    prep_convw_kernel<<<g1(64 * 32), 256, 0, stream>>>(b1_t1w, wB1T1, 64, 2, 3, 32);
    prep_convw_kernel<<<g1(128 * 96), 256, 0, stream>>>(b1_t2w, wB1T2, 128, 32, 3, 96);
    prep_convw_kernel<<<g1(128 * 192), 256, 0, stream>>>(b2_t1w, wB2T1, 128, 64, 3, 192);
    prep_convw_kernel<<<g1(256 * 192), 256, 0, stream>>>(b2_t2w, wB2T2, 256, 64, 3, 192);
    prep_convw_kernel<<<g1(256 * 5376), 256, 0, stream>>>(o_t1w, wOT1, 256, 128, 42, 5376);
    prep_convw_kernel<<<g1(32 * 32), 256, 0, stream>>>(b1_t1a, aB1T1, 32, 2, 1, 32);
    prep_convw_kernel<<<g1(64 * 32), 256, 0, stream>>>(b1_t2a, aB1T2, 64, 32, 1, 32);
    prep_convw_kernel<<<g1(128 * 64), 256, 0, stream>>>(b2_t2a, aB2T2, 128, 64, 1, 64);
    prep_convw_kernel<<<g1(128 * 128), 256, 0, stream>>>(o_t2w, wOT2, 128, 128, 1, 128);
    prep_convw_kernel<<<g1(20 * 128), 256, 0, stream>>>(o_fw, wOF, 20, 128, 1, 128);
    prep_th_kernel<<<g1(3 * 32 * 32), 256, 0, stream>>>(b1_th, th1, 32, 32, 32);
    prep_th_kernel<<<g1(3 * 64 * 64), 256, 0, stream>>>(b2_th, th2, 64, 64, 64);

    // ---- input transpose + graph basis ----
    transpose_in_kernel<<<g1(2 * 50 * NNODES * BB), 256, 0, stream>>>(x, A0);
    cheb_kernel<<<dim3(BB), 64, 0, stream>>>(x, Lk);

    // ---- block 1 ----
    tconv_glu_kernel<<<dim3(64, 48 * NNODES), 64, 0, stream>>>(
        wB1T1, 32, aB1T1, 32, b1_t1b, A0, arA, 2, 32, 3, 50, 48, 2);
    for (int t0 = 0; t0 < 48; t0 += 12) {
        theta_gemm_kernel<<<dim3(64, 12 * NNODES, 3), 64, 0, stream>>>(
            th1, arA, uA, 32, 32, 32, 48, t0, 12);
        sconv_step2_kernel<<<dim3(12, 64), 256, 0, stream>>>(
            uA, Lk, arA, b1_sb, arB, 32, 48, t0, 12);
    }
    tconv_glu_kernel<<<dim3(64, 46 * NNODES), 128, 0, stream>>>(
        wB1T2, 96, aB1T2, 32, b1_t2b, arB, arA, 32, 64, 3, 48, 46, 32);
    ln_kernel<<<dim3(46), 1024, 0, stream>>>(arA, b1_g, b1_bb, 64, 46);

    // ---- block 2 ----
    tconv_glu_kernel<<<dim3(64, 44 * NNODES), 128, 0, stream>>>(
        wB2T1, 192, nullptr, 0, b2_t1b, arA, arB, 64, 64, 3, 46, 44, 64);
    for (int t0 = 0; t0 < 44; t0 += 11) {
        theta_gemm_kernel<<<dim3(64, 11 * NNODES, 3), 128, 0, stream>>>(
            th2, arB, uA, 64, 64, 64, 44, t0, 11);
        sconv_step2_kernel<<<dim3(11, 64), 256, 0, stream>>>(
            uA, Lk, arB, b2_sb, arA, 64, 44, t0, 11);
    }
    tconv_glu_kernel<<<dim3(64, 42 * NNODES), 256, 0, stream>>>(
        wB2T2, 192, aB2T2, 64, b2_t2b, arA, arB, 64, 128, 3, 44, 42, 64);
    ln_kernel<<<dim3(42), 1024, 0, stream>>>(arB, b2_g, b2_bb, 128, 42);

    // ---- output stage ----
    tconv_glu_kernel<<<dim3(64, 1 * NNODES), 256, 0, stream>>>(
        wOT1, 5376, nullptr, 0, o_t1b, arB, arA, 128, 128, 42, 42, 1, 32);
    ln_kernel<<<dim3(1), 1024, 0, stream>>>(arA, o_g, o_bb, 128, 1);
    final_kernel<<<dim3(NNODES, 4), 256, 0, stream>>>(
        arA, wOT2, o_t2b, wOF, o_fb, (float*)d_out);
}
